// TorchInformerModel_5832565588138
// MI455X (gfx1250) — matrix operand order repaired and checked by reading, not compiled, not run on hardware
//
#include <hip/hip_runtime.h>
#include <cmath>

// ---------------------------------------------------------------------------
// CDNA5 (gfx1250) Informer forward pass.
//  - All GEMMs + both attention matmuls on v_wmma_f32_16x16x32_bf16 (wave32).
//  - GEMM: 64x128x32 block tile, double-buffered LDS, batched global loads.
//  - Q/K/V activations are bf16 (produced by GEMM epilogue); flash attention
//    stages K tiles with GLOBAL_LOAD_ASYNC_TO_LDS (ASYNCcnt) when available.
// ---------------------------------------------------------------------------

typedef __attribute__((ext_vector_type(16))) __bf16 v16bf;
typedef __attribute__((ext_vector_type(8)))  __bf16 v8bf;
typedef __attribute__((ext_vector_type(8)))  float  v8f;
typedef int v4i_ls __attribute__((vector_size(16)));   // matches builtin param

#define AS1 __attribute__((address_space(1)))
#define AS3 __attribute__((address_space(3)))

#if __has_builtin(__builtin_amdgcn_global_load_async_to_lds_b128)
#define HAVE_ASYNC_LDS 1
#else
#define HAVE_ASYNC_LDS 0
#endif

__device__ __forceinline__ void wait_asynccnt0() {
#if __has_builtin(__builtin_amdgcn_s_wait_asynccnt)
  __builtin_amdgcn_s_wait_asynccnt(0);
#else
  asm volatile("s_wait_asynccnt 0" ::: "memory");
#endif
}

#define BATCH    64
#define SEQ_LEN  384
#define DEC_LEN  192
#define PRED_LEN 96
#define DMODEL   512
#define NHEADS   8
#define HDIM     64
#define DFF      2048

__device__ __forceinline__ v8f wmma_bf16(v16bf a, v16bf b, v8f c) {
  return __builtin_amdgcn_wmma_f32_16x16x32_bf16(false, a, false, b,
                                                 (short)0, c, false, false);
}

// ---------------------------------------------------------------------------
// Embedding: out[b,l,d] = x@Wv + b_v + xm@Wt + b_t + PE[l,d]
// ---------------------------------------------------------------------------
__global__ __launch_bounds__(256) void embed_kernel(
    const float* __restrict__ x, const float* __restrict__ xm,
    const float* __restrict__ Wv, const float* __restrict__ bv,
    const float* __restrict__ Wt, const float* __restrict__ bt,
    float* __restrict__ out, int L, int cin, int tin)
{
  int idx = blockIdx.x * 256 + threadIdx.x;        // over B*L*512
  int d  = idx & (DMODEL - 1);
  int rl = idx >> 9;                                // b*L + l
  int l  = rl % L;
  float s = bv[d] + bt[d];
  const float* xr = x + (size_t)rl * cin;
  for (int c = 0; c < cin; c++) s += xr[c] * Wv[c * DMODEL + d];
  const float* mr = xm + (size_t)rl * tin;
  for (int c = 0; c < tin; c++) s += mr[c] * Wt[c * DMODEL + d];
  int i2 = d & ~1;
  float div = __expf(-(float)i2 * (9.210340371976184f / (float)DMODEL)); // ln(1e4)
  float ang = (float)l * div;
  s += (d & 1) ? __cosf(ang) : __sinf(ang);
  out[idx] = s;
}

// ---------------------------------------------------------------------------
// Tiled bf16 WMMA GEMM: C = act(A[M,K] @ W[K,N] + bias), f32 or bf16 output.
// Block tile 64x128, BK=32, 8 waves (2x4), wave tile 32x32 (2x2 wmma frags).
// Double-buffered LDS: next tile's global loads overlap current WMMAs.
// ---------------------------------------------------------------------------
#define BM  64
#define BN  128
#define BK  32
#define LDA 48   // bf16 elems; 96B row stride keeps 32B-aligned vector ds loads
#define LDB 48

__global__ __launch_bounds__(256) void gemm_kernel(
    const float* __restrict__ A, const float* __restrict__ W,
    const float* __restrict__ bias, float* __restrict__ C,
    __bf16* __restrict__ Cb, int M, int N, int K, int act, int obf)
{
  __shared__ __bf16 As[2][BM * LDA];
  __shared__ __bf16 Bs[2][BN * LDB];
  const int tid  = threadIdx.x;
  const int wid  = tid >> 5;
  const int lane = tid & 31;
  const int half = lane >> 4;
  const int l16  = lane & 15;
  const int m0 = blockIdx.y * BM;
  const int n0 = blockIdx.x * BN;
  const int wm = (wid >> 2) * 32;
  const int wn = (wid & 3) * 32;

  v8f acc[2][2];
  for (int i = 0; i < 2; i++)
    for (int j = 0; j < 2; j++)
      acc[i][j] = (v8f){0, 0, 0, 0, 0, 0, 0, 0};

  // batched tile loads: issue all global_load_b128 before any use
  float4 fa[2], fb[4];
  const int ar  = (tid * 2) >> 3;          // A row for this thread (2 chunks)
  const int ac4 = ((tid * 2) & 7) * 4;     // first f32 col of chunk 0
  auto load_tile = [&](int k0) {
    for (int i = 0; i < 2; i++) {
      int e  = tid + i * 256;              // 512 float4 chunks of A (64x32)
      int r  = e >> 3;
      int c4 = (e & 7) * 4;
      fa[i] = *(const float4*)(A + (size_t)(m0 + r) * K + k0 + c4);
    }
    for (int i = 0; i < 4; i++) {
      int e  = tid + i * 256;              // 1024 float4 chunks of W (32x128)
      int kk = e >> 5;
      int n4 = (e & 31) * 4;
      fb[i] = *(const float4*)(W + (size_t)(k0 + kk) * N + n0 + n4);
    }
  };
  auto store_tile = [&](int buf) {
    for (int i = 0; i < 2; i++) {
      int e  = tid + i * 256;
      int r  = e >> 3;
      int c4 = (e & 7) * 4;
      __bf16* dst = &As[buf][r * LDA + c4];
      dst[0] = (__bf16)fa[i].x; dst[1] = (__bf16)fa[i].y;
      dst[2] = (__bf16)fa[i].z; dst[3] = (__bf16)fa[i].w;
    }
    for (int i = 0; i < 4; i++) {
      int e  = tid + i * 256;
      int kk = e >> 5;
      int n4 = (e & 31) * 4;
      Bs[buf][(n4 + 0) * LDB + kk] = (__bf16)fb[i].x;
      Bs[buf][(n4 + 1) * LDB + kk] = (__bf16)fb[i].y;
      Bs[buf][(n4 + 2) * LDB + kk] = (__bf16)fb[i].z;
      Bs[buf][(n4 + 3) * LDB + kk] = (__bf16)fb[i].w;
    }
  };

  const int nk = K / BK;
  load_tile(0);
  store_tile(0);
  for (int kt = 0; kt < nk; kt++) {
    __syncthreads();
    if (kt + 1 < nk) load_tile((kt + 1) * BK);     // overlap with WMMA below
    if (kt + 2 < nk)                               // warm L2 for tile kt+2
      __builtin_prefetch(A + (size_t)(m0 + ar) * K + (kt + 2) * BK + ac4, 0, 1);
    const int cur = kt & 1;
    v16bf a[2], b[2];
    // A-frag (16x32 bf16): half 0 -> K{0..15}, half 1 -> K{16..31}
    for (int mi = 0; mi < 2; mi++) {
      const __bf16* base = &As[cur][(wm + mi * 16 + l16) * LDA];
      union { v16bf v; v8bf h[2]; } u;
      u.v = *(const v16bf*)(base + half * 16);
      a[mi] = u.v;
    }
    // B-frag (32x16 bf16): lane holds col l16, contiguous K chunk of 16, same k order as A
    for (int ni = 0; ni < 2; ni++)
      b[ni] = *(const v16bf*)&Bs[cur][(wn + ni * 16 + l16) * LDB + half * 16];
    for (int mi = 0; mi < 2; mi++)
      for (int ni = 0; ni < 2; ni++)
        acc[mi][ni] = wmma_bf16(a[mi], b[ni], acc[mi][ni]);
    if (kt + 1 < nk) store_tile((kt + 1) & 1);
  }

  // epilogue: C layout vgpr i -> row i + 8*half, col l16
  for (int mi = 0; mi < 2; mi++)
    for (int ni = 0; ni < 2; ni++)
      for (int i = 0; i < 8; i++) {
        int row = m0 + wm + mi * 16 + i + 8 * half;
        int col = n0 + wn + ni * 16 + l16;
        float v = acc[mi][ni][i] + bias[col];
        if (act == 1) v = 0.5f * v * (1.0f + erff(v * 0.70710678118654752f));
        if (obf) Cb[(size_t)row * N + col] = (__bf16)v;
        else     C [(size_t)row * N + col] = v;
      }
}

// ---------------------------------------------------------------------------
// Fused flash-style attention over bf16 Q/K/V. Grid: (Lq/32, B*H), block 64
// (2 waves), each wave owns 16 q rows. K tiles staged via async LDS copy.
// ---------------------------------------------------------------------------
#define KTS 80   // K tile row stride  (bf16): [32 keys][64 dims]
#define VTS 48   // V^T tile row stride (bf16): [64 dims][32 keys]

__global__ __launch_bounds__(64) void flash_attn_kernel(
    const __bf16* __restrict__ Qp, const __bf16* __restrict__ Kp,
    const __bf16* __restrict__ Vp, float* __restrict__ Op,
    int Lq, int Lk, int causal)
{
  __shared__ __bf16 Kt[32 * KTS];
  __shared__ __bf16 Vt[64 * VTS];
  __shared__ __bf16 Ps[2 * 16 * 32];
  const int tid  = threadIdx.x;
  const int wid  = tid >> 5;
  const int lane = tid & 31;
  const int half = lane >> 4;
  const int l16  = lane & 15;
  const int b = blockIdx.y >> 3;
  const int h = blockIdx.y & 7;
  const int q0 = blockIdx.x * 32 + wid * 16;

  // Q fragments (held for whole K loop): 16B bf16 vector loads from global
  const __bf16* Qr = Qp + ((size_t)(b * Lq + q0 + l16) * DMODEL + h * HDIM);
  v16bf aQ[2];
  for (int j = 0; j < 2; j++) {
    union { v16bf v; v8bf h2[2]; } u;
    u.v = *(const v16bf*)(Qr + j * 32 + half * 16);
    aQ[j] = u.v;
  }

  v8f acc[4];
  for (int j = 0; j < 4; j++) acc[j] = (v8f){0, 0, 0, 0, 0, 0, 0, 0};
  float mrow[8], srow[8];
  for (int i = 0; i < 8; i++) { mrow[i] = -1e30f; srow[i] = 0.0f; }

  int ntiles = Lk >> 5;
  if (causal) { int lim = blockIdx.x + 1; if (lim < ntiles) ntiles = lim; }

  for (int kt = 0; kt < ntiles; kt++) {
    __syncthreads();
    // ---- K tile [key][dim] : async bytes -> LDS (ASYNCcnt), no VGPR trip
#if HAVE_ASYNC_LDS
    for (int i = 0; i < 4; i++) {
      int e   = tid + i * 64;             // 256 chunks of 8 bf16 (16B)
      int key = e >> 3;
      int seg = (e & 7) * 8;
      const __bf16* g = Kp + (size_t)(b * Lk + kt * 32 + key) * DMODEL + h * HDIM + seg;
      __bf16* l = &Kt[key * KTS + seg];
      __builtin_amdgcn_global_load_async_to_lds_b128(
          (AS1 v4i_ls*)g, (AS3 v4i_ls*)l, 0, 0);
    }
#else
    {
      uint4 kk[4];
      for (int i = 0; i < 4; i++) {
        int e   = tid + i * 64;
        int key = e >> 3;
        int seg = (e & 7) * 8;
        kk[i] = *(const uint4*)(Kp + (size_t)(b * Lk + kt * 32 + key) * DMODEL + h * HDIM + seg);
      }
      for (int i = 0; i < 4; i++) {
        int e   = tid + i * 64;
        int key = e >> 3;
        int seg = (e & 7) * 8;
        union { uint4 u; v8bf v; } cv; cv.u = kk[i];
        *(v8bf*)&Kt[key * KTS + seg] = cv.v;
      }
    }
#endif
    // ---- V^T tile [dim][key] : batched bf16 loads, transposed LDS scatter
    {
      uint4 vv[4];
      for (int i = 0; i < 4; i++) {
        int e   = tid + i * 64;
        int key = e >> 3;
        int d8  = (e & 7) * 8;
        vv[i] = *(const uint4*)(Vp + (size_t)(b * Lk + kt * 32 + key) * DMODEL + h * HDIM + d8);
      }
      for (int i = 0; i < 4; i++) {
        int e   = tid + i * 64;
        int key = e >> 3;
        int d8  = (e & 7) * 8;
        union { uint4 u; __bf16 h2[8]; } cv; cv.u = vv[i];
        for (int j = 0; j < 8; j++) Vt[(d8 + j) * VTS + key] = cv.h2[j];
      }
    }
#if HAVE_ASYNC_LDS
    wait_asynccnt0();                      // barrier does not drain ASYNCcnt
#endif
    __syncthreads();

    // ---- scores: two 16x16 tiles, K-dim = head dim (2 x wmma K=32)
    float s[2][8];
    for (int t = 0; t < 2; t++) {
      v8f cs = (v8f){0, 0, 0, 0, 0, 0, 0, 0};
      for (int j = 0; j < 2; j++) {
        v16bf bk = *(const v16bf*)&Kt[(t * 16 + l16) * KTS + j * 32 + half * 16];
        cs = wmma_bf16(aQ[j], bk, cs);
      }
      int col = kt * 32 + t * 16 + l16;
      for (int i = 0; i < 8; i++) {
        float v = cs[i] * 0.125f;                 // 1/sqrt(64)
        int row = q0 + i + 8 * half;
        if (causal && col > row) v = -1.0e9f;
        s[t][i] = v;
      }
    }

    // ---- online softmax (row lives on 16 lanes of one wave-half)
    for (int i = 0; i < 8; i++) {
      float mloc = fmaxf(s[0][i], s[1][i]);
      for (int d = 1; d < 16; d <<= 1) mloc = fmaxf(mloc, __shfl_xor(mloc, d));
      float mnew  = fmaxf(mrow[i], mloc);
      float alpha = __expf(mrow[i] - mnew);
      mrow[i] = mnew;
      float p0 = __expf(s[0][i] - mnew);
      float p1 = __expf(s[1][i] - mnew);
      float ls = p0 + p1;
      for (int d = 1; d < 16; d <<= 1) ls += __shfl_xor(ls, d);
      srow[i] = srow[i] * alpha + ls;
      for (int j = 0; j < 4; j++) acc[j][i] *= alpha;
      int prow = i + 8 * half;
      Ps[wid * 512 + prow * 32 + l16]      = (__bf16)p0;
      Ps[wid * 512 + prow * 32 + 16 + l16] = (__bf16)p1;
    }

    // ---- P[16x32] @ V[32x64]  (LDS in-order within wave covers Ps RAW)
    const __bf16* pb = &Ps[wid * 512 + l16 * 32];
    union { v16bf v; v8bf hh[2]; } up;
    up.v = *(const v16bf*)(pb + half * 16);
    v16bf aP = up.v;
    for (int j = 0; j < 4; j++) {
      v16bf bv = *(const v16bf*)&Vt[(j * 16 + l16) * VTS + half * 16];
      acc[j] = wmma_bf16(aP, bv, acc[j]);
    }
  }

  for (int j = 0; j < 4; j++)
    for (int i = 0; i < 8; i++) {
      int row  = q0 + i + 8 * half;
      int dcol = j * 16 + l16;
      Op[(size_t)(b * Lq + row) * DMODEL + h * HDIM + dcol] = acc[j][i] / srow[i];
    }
}

// ---------------------------------------------------------------------------
// ProbSparse: top-u of ||q||^2 per (b,h) -> binary row mask
// ---------------------------------------------------------------------------
__global__ __launch_bounds__(256) void topk_mask_kernel(
    const __bf16* __restrict__ Q, float* __restrict__ mask, int Lq, int u)
{
  __shared__ float norms[SEQ_LEN];
  __shared__ float rv[256];
  __shared__ int   ri[256];
  int tid = threadIdx.x;
  int b = blockIdx.x >> 3;
  int h = blockIdx.x & 7;
  for (int l = tid; l < Lq; l += 256) {
    const __bf16* q = Q + ((size_t)(b * Lq + l) * DMODEL + h * HDIM);
    float s = 0.f;
    for (int d = 0; d < HDIM; d++) { float v = (float)q[d]; s += v * v; }
    norms[l] = s;
    mask[(size_t)blockIdx.x * Lq + l] = 0.0f;
  }
  __syncthreads();
  for (int it = 0; it < u; it++) {
    float best = -1e30f; int bi = 0;
    for (int l = tid; l < Lq; l += 256)
      if (norms[l] > best) { best = norms[l]; bi = l; }
    rv[tid] = best; ri[tid] = bi;
    __syncthreads();
    for (int s = 128; s > 0; s >>= 1) {
      if (tid < s && rv[tid + s] > rv[tid]) { rv[tid] = rv[tid + s]; ri[tid] = ri[tid + s]; }
      __syncthreads();
    }
    if (tid == 0) {
      norms[ri[0]] = -1e30f;
      mask[(size_t)blockIdx.x * Lq + ri[0]] = 1.0f;
    }
    __syncthreads();
  }
}

__global__ __launch_bounds__(256) void apply_mask_kernel(
    float* __restrict__ ctx, const float* __restrict__ mask, int Lq)
{
  size_t idx = (size_t)blockIdx.x * 256 + threadIdx.x;  // B*Lq*512
  int d  = (int)(idx & (DMODEL - 1));
  int rl = (int)(idx >> 9);
  int b = rl / Lq, l = rl % Lq;
  int h = d >> 6;
  ctx[idx] *= mask[(size_t)(b * NHEADS + h) * Lq + l];
}

// ---------------------------------------------------------------------------
// LayerNorm(x [+ res]) * g + b   (one 128-thread block per 512-wide row)
// ---------------------------------------------------------------------------
__global__ __launch_bounds__(128) void ln_kernel(
    const float* __restrict__ x, const float* __restrict__ res,
    const float* __restrict__ g, const float* __restrict__ bb,
    float* __restrict__ out)
{
  __shared__ float red[128];
  int row = blockIdx.x, tid = threadIdx.x;
  size_t base = (size_t)row * DMODEL;
  float v[4];
  for (int i = 0; i < 4; i++) {
    int d = tid + i * 128;
    float t = x[base + d];
    if (res) t += res[base + d];
    v[i] = t;
  }
  red[tid] = v[0] + v[1] + v[2] + v[3];
  __syncthreads();
  for (int st = 64; st > 0; st >>= 1) { if (tid < st) red[tid] += red[tid + st]; __syncthreads(); }
  float mu = red[0] * (1.0f / (float)DMODEL);
  __syncthreads();
  float q = 0.f;
  for (int i = 0; i < 4; i++) { float d0 = v[i] - mu; q += d0 * d0; }
  red[tid] = q;
  __syncthreads();
  for (int st = 64; st > 0; st >>= 1) { if (tid < st) red[tid] += red[tid + st]; __syncthreads(); }
  float rstd = rsqrtf(red[0] * (1.0f / (float)DMODEL) + 1e-5f);
  for (int i = 0; i < 4; i++) {
    int d = tid + i * 128;
    out[base + d] = (v[i] - mu) * rstd * g[d] + bb[d];
  }
}

// ---------------------------------------------------------------------------
// Distil maxpool: window 3, stride 2, pad 1
// ---------------------------------------------------------------------------
__global__ __launch_bounds__(256) void pool_kernel(
    const float* __restrict__ x, float* __restrict__ out, int Lin, int Lout)
{
  size_t idx = (size_t)blockIdx.x * 256 + threadIdx.x;  // B*Lout*512
  int d  = (int)(idx & (DMODEL - 1));
  int rl = (int)(idx >> 9);
  int b = rl / Lout, lo = rl % Lout;
  float m = -3.0e38f;
  for (int t = -1; t <= 1; t++) {
    int li = 2 * lo + t;
    if (li >= 0 && li < Lin)
      m = fmaxf(m, x[((size_t)(b * Lin + li) << 9) + d]);
  }
  out[idx] = m;
}

// ---------------------------------------------------------------------------
// Final projection 512 -> 7, only last PRED_LEN rows of each batch
// ---------------------------------------------------------------------------
__global__ __launch_bounds__(256) void proj_kernel(
    const float* __restrict__ dec, const float* __restrict__ Wp,
    const float* __restrict__ bp, float* __restrict__ out)
{
  int idx = blockIdx.x * 256 + threadIdx.x;     // 64*96*7
  int c  = idx % 7;
  int bt = idx / 7;
  int b = bt / PRED_LEN, t = bt % PRED_LEN;
  size_t rbase = (size_t)(b * DEC_LEN + (DEC_LEN - PRED_LEN) + t) * DMODEL;
  float s = bp[c];
  for (int d = 0; d < DMODEL; d++) s += dec[rbase + d] * Wp[d * 7 + c];
  out[idx] = s;
}

// ---------------------------------------------------------------------------
// Host orchestration
// ---------------------------------------------------------------------------
extern "C" void kernel_launch(void* const* d_in, const int* in_sizes, int n_in,
                              void* d_out, int out_size, void* d_ws, size_t ws_size,
                              hipStream_t stream)
{
  (void)in_sizes; (void)n_in; (void)out_size; (void)ws_size;

  const float* x_enc      = (const float*)d_in[0];
  const float* x_mark_enc = (const float*)d_in[1];
  const float* x_dec      = (const float*)d_in[2];
  const float* x_mark_dec = (const float*)d_in[3];

  struct Lin  { const float *w, *b; };
  struct Norm { const float *g, *b; };
  struct AttnP { Lin q, k, v, o; };
  struct EncL { AttnP a; Lin f1, f2; Norm n1, n2; };
  struct DecL { AttnP s, c; Lin f1, f2; Norm n1, n2, n3; };

  int pi = 4;
  auto P   = [&]() { return (const float*)d_in[pi++]; };
  auto lin = [&]() { Lin l; l.w = P(); l.b = P(); return l; };
  auto nrm = [&]() { Norm n; n.g = P(); n.b = P(); return n; };

  Lin enc_val = lin(), enc_tmp = lin();
  Lin dec_val = lin(), dec_tmp = lin();
  EncL enc[3];
  for (int i = 0; i < 3; i++) {
    enc[i].a.q = lin(); enc[i].a.k = lin(); enc[i].a.v = lin(); enc[i].a.o = lin();
    enc[i].f1 = lin(); enc[i].f2 = lin();
    enc[i].n1 = nrm(); enc[i].n2 = nrm();
  }
  Norm distil = nrm();
  DecL dcl[2];
  for (int i = 0; i < 2; i++) {
    dcl[i].s.q = lin(); dcl[i].s.k = lin(); dcl[i].s.v = lin(); dcl[i].s.o = lin();
    dcl[i].c.q = lin(); dcl[i].c.k = lin(); dcl[i].c.v = lin(); dcl[i].c.o = lin();
    dcl[i].f1 = lin(); dcl[i].f2 = lin();
    dcl[i].n1 = nrm(); dcl[i].n2 = nrm(); dcl[i].n3 = nrm();
  }
  Lin prj = lin();

  // workspace carve-up (SZ = one [B,384,512] f32 tensor = 50.3 MB)
  char* ws = (char*)d_ws;
  const size_t SZ = (size_t)BATCH * SEQ_LEN * DMODEL * sizeof(float);
  float*  X    = (float*)(ws + 0 * SZ);
  float*  CTX  = (float*)(ws + 1 * SZ);
  float*  AA   = (float*)(ws + 2 * SZ);
  float*  Hb   = (float*)(ws + 3 * SZ);            // spans 4*SZ (B*384*2048 f32)
  float*  Db   = (float*)(ws + 7 * SZ);            // decoder activations (SZ/2)
  __bf16* Qb   = (__bf16*)(ws + 7 * SZ + SZ / 2);  // bf16 Q/K/V, SZ/2 each
  __bf16* Kb   = (__bf16*)(ws + 8 * SZ);
  __bf16* Vb   = (__bf16*)(ws + 8 * SZ + SZ / 2);
  float*  MASK = (float*)(ws + 9 * SZ);

  auto gemm = [&](const float* A, const Lin& w, float* Cf, __bf16* Cbf,
                  int M, int N, int K, int act) {
    gemm_kernel<<<dim3(N / BN, M / BM), 256, 0, stream>>>(
        A, w.w, w.b, Cf, Cbf, M, N, K, act, Cbf != nullptr);
  };
  auto lnrm = [&](const float* x, const float* r, const Norm& n, float* o, int rows) {
    ln_kernel<<<rows, 128, 0, stream>>>(x, r, n.g, n.b, o);
  };

  // -------------------- encoder --------------------
  embed_kernel<<<(BATCH * SEQ_LEN * DMODEL) / 256, 256, 0, stream>>>(
      x_enc, x_mark_enc, enc_val.w, enc_val.b, enc_tmp.w, enc_tmp.b,
      X, SEQ_LEN, 7, 4);

  int L = SEQ_LEN;
  for (int il = 0; il < 3; il++) {
    int M = BATCH * L;
    gemm(X, enc[il].a.q, nullptr, Qb, M, DMODEL, DMODEL, 0);
    gemm(X, enc[il].a.k, nullptr, Kb, M, DMODEL, DMODEL, 0);
    gemm(X, enc[il].a.v, nullptr, Vb, M, DMODEL, DMODEL, 0);
    flash_attn_kernel<<<dim3(L / 32, BATCH * NHEADS), 64, 0, stream>>>(
        Qb, Kb, Vb, CTX, L, L, 0);
    int u = (int)(5.0 * std::log((double)L)); if (u < 1) u = 1;
    topk_mask_kernel<<<BATCH * NHEADS, 256, 0, stream>>>(Qb, MASK, L, u);
    apply_mask_kernel<<<(unsigned)((size_t)M * DMODEL / 256), 256, 0, stream>>>(CTX, MASK, L);
    gemm(CTX, enc[il].a.o, AA, nullptr, M, DMODEL, DMODEL, 0);
    lnrm(X, AA, enc[il].n1, X, M);
    gemm(X, enc[il].f1, Hb, nullptr, M, DFF, DMODEL, 1);
    gemm(Hb, enc[il].f2, AA, nullptr, M, DMODEL, DFF, 0);
    lnrm(X, AA, enc[il].n2, X, M);
    if (il < 2) {
      int Lo = L / 2;
      pool_kernel<<<(BATCH * Lo * DMODEL) / 256, 256, 0, stream>>>(X, CTX, L, Lo);
      lnrm(CTX, nullptr, distil, X, BATCH * Lo);
      L = Lo;
    }
  }
  const int enc_valid = L;  // 96; X holds enc_out [B,96,512]

  // -------------------- decoder --------------------
  embed_kernel<<<(BATCH * DEC_LEN * DMODEL) / 256, 256, 0, stream>>>(
      x_dec, x_mark_dec, dec_val.w, dec_val.b, dec_tmp.w, dec_tmp.b,
      Db, DEC_LEN, 7, 4);

  const int Md = BATCH * DEC_LEN;
  const int Me = BATCH * enc_valid;
  for (int il = 0; il < 2; il++) {
    // masked self-attention
    gemm(Db, dcl[il].s.q, nullptr, Qb, Md, DMODEL, DMODEL, 0);
    gemm(Db, dcl[il].s.k, nullptr, Kb, Md, DMODEL, DMODEL, 0);
    gemm(Db, dcl[il].s.v, nullptr, Vb, Md, DMODEL, DMODEL, 0);
    flash_attn_kernel<<<dim3(DEC_LEN / 32, BATCH * NHEADS), 64, 0, stream>>>(
        Qb, Kb, Vb, CTX, DEC_LEN, DEC_LEN, 1);
    gemm(CTX, dcl[il].s.o, AA, nullptr, Md, DMODEL, DMODEL, 0);
    lnrm(Db, AA, dcl[il].n1, Db, Md);
    // cross-attention against encoder output
    gemm(Db, dcl[il].c.q, nullptr, Qb, Md, DMODEL, DMODEL, 0);
    gemm(X,  dcl[il].c.k, nullptr, Kb, Me, DMODEL, DMODEL, 0);
    gemm(X,  dcl[il].c.v, nullptr, Vb, Me, DMODEL, DMODEL, 0);
    flash_attn_kernel<<<dim3(DEC_LEN / 32, BATCH * NHEADS), 64, 0, stream>>>(
        Qb, Kb, Vb, CTX, DEC_LEN, enc_valid, 0);
    gemm(CTX, dcl[il].c.o, AA, nullptr, Md, DMODEL, DMODEL, 0);
    lnrm(Db, AA, dcl[il].n2, Db, Md);
    // FFN
    gemm(Db, dcl[il].f1, Hb, nullptr, Md, DFF, DMODEL, 1);
    gemm(Hb, dcl[il].f2, AA, nullptr, Md, DMODEL, DFF, 0);
    lnrm(Db, AA, dcl[il].n3, Db, Md);
  }

  proj_kernel<<<(BATCH * PRED_LEN * 7) / 256, 256, 0, stream>>>(
      Db, prj.w, prj.b, (float*)d_out);
}
